// HeadAttention_63282048139412
// MI455X (gfx1250) — matrix operand order repaired and checked by reading, not compiled, not run on hardware
//
#include <hip/hip_runtime.h>
#include <hip/hip_bf16.h>
#include <math.h>

// ---------------------------------------------------------------------------
// HeadAttention for MI455X (gfx1250): all GEMMs via v_wmma_f32_16x16x32_f16,
// all f16 tile staging via double-buffered global_load_async_to_lds_b128.
//   qh = q Wq^T ; kh = k Wk^T ; vh = v Wv^T   (f16 in workspace)
//   S = qh kh^T * 8^-1, causal softmax (two-pass: stats then recompute)
//   y = P vh (f16) ; out = y Wproj^T (f32)
// Workspace (~72.1 MiB): 4x f16 weights (8Mi) | qh,kh,vh,y f16 (64Mi) | stats.
// ---------------------------------------------------------------------------

typedef _Float16 v16h __attribute__((ext_vector_type(16)));
typedef _Float16 v4h  __attribute__((ext_vector_type(4)));
typedef float    v8f  __attribute__((ext_vector_type(8)));
typedef float    f32x4 __attribute__((ext_vector_type(4)));

#define DM    1024
#define SEQ   2048
#define BATCH 4
#define SCALE 0.125f          // 1/sqrt(64)
#define NEGBIG (-1.0e30f)

union FragU { v16h v; _Float16 h[16]; f32x4 f4[2]; };

// ------------------------- CDNA5 async global->LDS -------------------------
__device__ __forceinline__ uint32_t lds_off(const void* p) {
  return (uint32_t)(uintptr_t)p;   // LDS aperture: byte offset lives in addr[31:0]
}

// GVS form: mem = SGPR64 base + per-lane VGPR32 offset; LDS dest from VGPR.
__device__ __forceinline__ void async_b128(uint32_t ldsoff, uint32_t voff, uint64_t sbase) {
  asm volatile("global_load_async_to_lds_b128 %0, %1, %2"
               :: "v"(ldsoff), "v"(voff), "s"(sbase) : "memory");
}

template <int N>
__device__ __forceinline__ void wait_async() {
  asm volatile("s_wait_asynccnt %0" :: "n"(N));
}

// Fill LDS tile [128][32] f16 (row pitch 32) from global f16 matrix with row
// pitch DM halves, starting column kc. 2 async instructions per thread.
__device__ __forceinline__ void fill_tile_async(uint32_t lds_base,
                                                const _Float16* g /*uniform: row0 of tile*/,
                                                int kc) {
  uint64_t sbase = (uint64_t)(uintptr_t)g + (uint64_t)(kc * 2);
  for (int t = threadIdx.x; t < 512; t += 256) {
    int r = t >> 2, c8 = (t & 3) << 3;                       // 8-half (16B) chunks
    async_b128(lds_base + (uint32_t)(r * 32 + c8) * 2,
               (uint32_t)(r * (DM * 2) + c8 * 2), sbase);
  }
}

// ------------------------------ WMMA fragments -----------------------------
// A-fragment (16x32 f16, MxK): lane&15 = row; lane half selects K {0..7,16..23}
// vs {8..15,24..31}.
__device__ __forceinline__ v16h frag_a(const _Float16* base, int row0, int k0, int ldk) {
  const int lane = threadIdx.x & 31;
  const _Float16* p = base + (row0 + (lane & 15)) * ldk + k0 + ((lane >> 4) << 3);
  FragU u;
  u.f4[0] = *(const f32x4*)(p);
  u.f4[1] = *(const f32x4*)(p + 16);
  return u.v;
}

// B-fragment (32x16 f16, KxN) from tile stored [n][k]: lane&15 = column n;
// lane half selects K {0..7,16..23} vs {8..15,24..31}.
__device__ __forceinline__ v16h frag_b(const _Float16* base, int col0, int k0, int ldk) {
  const int lane = threadIdx.x & 31;
  const _Float16* p = base + (col0 + (lane & 15)) * ldk + k0 + ((lane >> 4) << 3);
  FragU u;
  u.f4[0] = *(const f32x4*)(p);
  u.f4[1] = *(const f32x4*)(p + 16);
  return u.v;
}

__device__ __forceinline__ v8f wmma(v16h a, v16h b, v8f c) {
  return __builtin_amdgcn_wmma_f32_16x16x32_f16(false, a, false, b, (short)0, c, false, false);
}

// ---------------------------------------------------------------------------
// f32 -> f16 weight conversion (vectorized x4)
// ---------------------------------------------------------------------------
__global__ void cvt_f32_f16(const float* __restrict__ src, _Float16* __restrict__ dst, int n4) {
  int i = blockIdx.x * blockDim.x + threadIdx.x;
  if (i < n4) {
    f32x4 f = *(const f32x4*)(src + 4 * (size_t)i);
    v4h h;
    h[0] = (_Float16)f[0]; h[1] = (_Float16)f[1];
    h[2] = (_Float16)f[2]; h[3] = (_Float16)f[3];
    *(v4h*)(dst + 4 * (size_t)i) = h;
  }
}

// ---------------------------------------------------------------------------
// C[M x 1024] = A[M x 1024] * W^T (W stored [1024 x 1024] row-major, f16).
// Block tile 128x128, 8 waves; double-buffered K-chunks of 32 with async LDS.
// ---------------------------------------------------------------------------
template <typename AT, typename OT>
__global__ __launch_bounds__(256)
void gemm_xwt(const AT* __restrict__ A, const _Float16* __restrict__ W,
              OT* __restrict__ C) {
  __shared__ __align__(16) _Float16 As[2][128 * 32];
  __shared__ __align__(16) _Float16 Bs[2][128 * 32];
  const int tm = blockIdx.x * 128;
  const int tn = blockIdx.y * 128;
  const int w    = threadIdx.x >> 5;
  const int lane = threadIdx.x & 31;
  const int hi   = lane >> 4;

  constexpr bool AF32 = (sizeof(AT) == 4);
  constexpr int  NASY = AF32 ? 2 : 4;       // async ops per thread per fill

  v8f acc[8] = {};

  auto fill = [&](int buf, int kc) {
    if constexpr (AF32) {
      for (int t = threadIdx.x; t < 1024; t += 256) {      // f32 -> f16 via VALU
        int r = t >> 3, c4 = (t & 7) << 2;
        f32x4 f = *(const f32x4*)(A + (size_t)(tm + r) * DM + kc + c4);
        v4h h;
        h[0] = (_Float16)f[0]; h[1] = (_Float16)f[1];
        h[2] = (_Float16)f[2]; h[3] = (_Float16)f[3];
        *(v4h*)(&As[buf][r * 32 + c4]) = h;
      }
    } else {
      fill_tile_async(lds_off(&As[buf][0]), (const _Float16*)A + (size_t)tm * DM, kc);
    }
    fill_tile_async(lds_off(&Bs[buf][0]), W + (size_t)tn * DM, kc);
  };

  fill(0, 0);
  for (int i = 0; i < DM / 32; ++i) {
    const int cur = i & 1;
    const bool more = (i + 1 < DM / 32);
    if (more) fill(1 - cur, (i + 1) * 32);
    if (more) wait_async<NASY>(); else wait_async<0>();
    __syncthreads();
    v16h a = frag_a(As[cur], w * 16, 0, 32);
#pragma unroll
    for (int j = 0; j < 8; ++j) {
      v16h b = frag_b(Bs[cur], j * 16, 0, 32);
      acc[j] = wmma(a, b, acc[j]);
    }
    __syncthreads();
  }

  const int rowb = tm + w * 16 + hi * 8;
  const int colb = tn + (lane & 15);
#pragma unroll
  for (int j = 0; j < 8; ++j)
#pragma unroll
    for (int r = 0; r < 8; ++r) {
      float v = acc[j][r];
      if constexpr (sizeof(OT) == 2)
        C[(size_t)(rowb + r) * DM + colb + j * 16] = (OT)(_Float16)v;
      else
        C[(size_t)(rowb + r) * DM + colb + j * 16] = (OT)v;
    }
}

// ---------------------------------------------------------------------------
// Pass 1: causal softmax statistics (row max m, row sum l) per query row.
// Block = 128 query rows (8 waves x 16 rows); key tiles of 128; async staging.
// ---------------------------------------------------------------------------
__global__ __launch_bounds__(256)
void attn_stats(const _Float16* __restrict__ qh, const _Float16* __restrict__ kh,
                float* __restrict__ mws, float* __restrict__ lws) {
  __shared__ __align__(16) _Float16 Qs[2][128 * 32];
  __shared__ __align__(16) _Float16 Ks[2][128 * 32];
  const int qt = blockIdx.x;
  const int b  = blockIdx.y;
  const int q0 = qt * 128;
  const int w    = threadIdx.x >> 5;
  const int lane = threadIdx.x & 31;
  const int hi   = lane >> 4;
  const size_t boff = (size_t)b * SEQ * DM;
  const _Float16* qrow0 = qh + boff + (size_t)q0 * DM;

  float rm[8], rl[8];
#pragma unroll
  for (int r = 0; r < 8; ++r) { rm[r] = NEGBIG; rl[r] = 0.f; }

  for (int kt = 0; kt <= qt; ++kt) {
    const int kb = kt * 128;
    const _Float16* krow0 = kh + boff + (size_t)kb * DM;
    v8f s[8] = {};

    auto fill = [&](int buf, int kc) {
      fill_tile_async(lds_off(&Qs[buf][0]), qrow0, kc);
      fill_tile_async(lds_off(&Ks[buf][0]), krow0, kc);
    };

    fill(0, 0);
    for (int i = 0; i < DM / 32; ++i) {
      const int cur = i & 1;
      const bool more = (i + 1 < DM / 32);
      if (more) fill(1 - cur, (i + 1) * 32);
      if (more) wait_async<4>(); else wait_async<0>();
      __syncthreads();
      v16h a = frag_a(Qs[cur], w * 16, 0, 32);
#pragma unroll
      for (int j = 0; j < 8; ++j) {
        v16h bf = frag_b(Ks[cur], j * 16, 0, 32);
        s[j] = wmma(a, bf, s[j]);
      }
      __syncthreads();
    }

    const bool diag = (kt == qt);
#pragma unroll
    for (int j = 0; j < 8; ++j) {
      int key = kb + j * 16 + (lane & 15);
#pragma unroll
      for (int r = 0; r < 8; ++r) {
        float v = s[j][r] * SCALE;
        if (diag) {
          int qrow = q0 + w * 16 + hi * 8 + r;
          if (key > qrow) v = NEGBIG;
        }
        s[j][r] = v;
      }
    }
#pragma unroll
    for (int r = 0; r < 8; ++r) {
      float tmax = NEGBIG;
#pragma unroll
      for (int j = 0; j < 8; ++j) tmax = fmaxf(tmax, s[j][r]);
#pragma unroll
      for (int xm = 1; xm < 16; xm <<= 1) tmax = fmaxf(tmax, __shfl_xor(tmax, xm, 32));
      float nm = fmaxf(rm[r], tmax);
      float psum = 0.f;
#pragma unroll
      for (int j = 0; j < 8; ++j) psum += __expf(s[j][r] - nm);
#pragma unroll
      for (int xm = 1; xm < 16; xm <<= 1) psum += __shfl_xor(psum, xm, 32);
      rl[r] = rl[r] * __expf(rm[r] - nm) + psum;
      rm[r] = nm;
    }
  }

  if ((lane & 15) == 0) {
    int base = b * SEQ + q0 + w * 16 + hi * 8;
#pragma unroll
    for (int r = 0; r < 8; ++r) { mws[base + r] = rm[r]; lws[base + r] = rl[r]; }
  }
}

// ---------------------------------------------------------------------------
// Pass 2: y[:, c0:c0+128] = softmax(S) * vh[:, c0:c0+128], scores recomputed.
// Grid (qTile, chanBlock, batch). P bounced through LDS for C->A relayout.
// ---------------------------------------------------------------------------
__global__ __launch_bounds__(256)
void attn_out(const _Float16* __restrict__ qh, const _Float16* __restrict__ kh,
              const _Float16* __restrict__ vh,
              const float* __restrict__ mws, const float* __restrict__ lws,
              _Float16* __restrict__ y) {
  __shared__ __align__(16) _Float16 P[128 * 128];                 // [q][k]
  __shared__ __align__(16) union ShU {
    _Float16 qk[2][2][128 * 32];   // [buf][0=Q,1=K] double-buffered score chunks
    _Float16 vt[128 * 128];        // V transposed [c][k]
  } sh;

  const int qt = blockIdx.x;
  const int c0 = blockIdx.y * 128;
  const int b  = blockIdx.z;
  const int q0 = qt * 128;
  const int w    = threadIdx.x >> 5;
  const int lane = threadIdx.x & 31;
  const int hi   = lane >> 4;
  const size_t boff = (size_t)b * SEQ * DM;
  const _Float16* qrow0 = qh + boff + (size_t)q0 * DM;

  float rm[8], rinv[8];
  {
    int base = b * SEQ + q0 + w * 16 + hi * 8;
#pragma unroll
    for (int r = 0; r < 8; ++r) { rm[r] = mws[base + r]; rinv[r] = 1.0f / lws[base + r]; }
  }

  v8f acc[8] = {};

  for (int kt = 0; kt <= qt; ++kt) {
    const int kb = kt * 128;
    const _Float16* krow0 = kh + boff + (size_t)kb * DM;
    v8f s[8] = {};

    auto fill = [&](int buf, int kc) {
      fill_tile_async(lds_off(&sh.qk[buf][0][0]), qrow0, kc);
      fill_tile_async(lds_off(&sh.qk[buf][1][0]), krow0, kc);
    };

    fill(0, 0);
    for (int i = 0; i < DM / 32; ++i) {
      const int cur = i & 1;
      const bool more = (i + 1 < DM / 32);
      if (more) fill(1 - cur, (i + 1) * 32);
      if (more) wait_async<4>(); else wait_async<0>();
      __syncthreads();
      v16h a = frag_a(sh.qk[cur][0], w * 16, 0, 32);
#pragma unroll
      for (int j = 0; j < 8; ++j) {
        v16h bf = frag_b(sh.qk[cur][1], j * 16, 0, 32);
        s[j] = wmma(a, bf, s[j]);
      }
      __syncthreads();
    }

    // P = exp(s*scale - m) / l  (zero above diagonal), write to LDS [q][k]
    const bool diag = (kt == qt);
#pragma unroll
    for (int j = 0; j < 8; ++j) {
      int key = kb + j * 16 + (lane & 15);
#pragma unroll
      for (int r = 0; r < 8; ++r) {
        int qrow = w * 16 + hi * 8 + r;
        float p = __expf(s[j][r] * SCALE - rm[r]) * rinv[r];
        if (diag && key > q0 + qrow) p = 0.f;
        P[qrow * 128 + j * 16 + (lane & 15)] = (_Float16)p;
      }
    }
    __syncthreads();   // P complete; qk buffers free

    // stage V tile transposed: vt[c][k] = vh[kb+k][c0+c]
    for (int t = threadIdx.x; t < 2048; t += 256) {
      int k = t >> 4, c8 = (t & 15) << 3;
      f32x4 d = *(const f32x4*)(vh + boff + (size_t)(kb + k) * DM + c0 + c8);
      const _Float16* hp = (const _Float16*)&d;
#pragma unroll
      for (int i = 0; i < 8; ++i) sh.vt[(c8 + i) * 128 + k] = hp[i];
    }
    __syncthreads();

    // Y += P * V   (K = 128 keys in 4 steps of 32)
    for (int ks = 0; ks < 128; ks += 32) {
      v16h a = frag_a(P, w * 16, ks, 128);
#pragma unroll
      for (int j = 0; j < 8; ++j) {
        v16h bf = frag_b(sh.vt, j * 16, ks, 128);
        acc[j] = wmma(a, bf, acc[j]);
      }
    }
    __syncthreads();   // protect vt/P before next iteration overwrites
  }

  const int rowb = q0 + w * 16 + hi * 8;
#pragma unroll
  for (int j = 0; j < 8; ++j)
#pragma unroll
    for (int r = 0; r < 8; ++r)
      y[boff + (size_t)(rowb + r) * DM + c0 + j * 16 + (lane & 15)] = (_Float16)acc[j][r];
}

// ---------------------------------------------------------------------------
extern "C" void kernel_launch(void* const* d_in, const int* in_sizes, int n_in,
                              void* d_out, int out_size, void* d_ws, size_t ws_size,
                              hipStream_t stream) {
  const float* q  = (const float*)d_in[0];
  const float* k  = (const float*)d_in[1];
  const float* v  = (const float*)d_in[2];
  const float* Wq = (const float*)d_in[3];
  const float* Wk = (const float*)d_in[4];
  const float* Wv = (const float*)d_in[5];
  const float* Wp = (const float*)d_in[6];
  float* out = (float*)d_out;

  const size_t WN = (size_t)DM * DM;            // 1M elements per weight
  const size_t XN = (size_t)BATCH * SEQ * DM;   // 8M elements per activation

  _Float16* wq16 = (_Float16*)d_ws;
  _Float16* wk16 = wq16 + WN;
  _Float16* wv16 = wk16 + WN;
  _Float16* wp16 = wv16 + WN;
  _Float16* qhp  = wp16 + WN;
  _Float16* khp  = qhp + XN;
  _Float16* vhp  = khp + XN;
  _Float16* yp   = vhp + XN;
  float*    mws  = (float*)(yp + XN);
  float*    lws  = mws + (size_t)BATCH * SEQ;

  const int n4 = (int)(WN / 4);
  cvt_f32_f16<<<dim3((n4 + 255) / 256), dim3(256), 0, stream>>>(Wq, wq16, n4);
  cvt_f32_f16<<<dim3((n4 + 255) / 256), dim3(256), 0, stream>>>(Wk, wk16, n4);
  cvt_f32_f16<<<dim3((n4 + 255) / 256), dim3(256), 0, stream>>>(Wv, wv16, n4);
  cvt_f32_f16<<<dim3((n4 + 255) / 256), dim3(256), 0, stream>>>(Wp, wp16, n4);

  dim3 blk(256);
  dim3 gg(64, 8);  // 8192/128 x 1024/128
  gemm_xwt<float, _Float16><<<gg, blk, 0, stream>>>(q, wq16, qhp);
  gemm_xwt<float, _Float16><<<gg, blk, 0, stream>>>(k, wk16, khp);
  gemm_xwt<float, _Float16><<<gg, blk, 0, stream>>>(v, wv16, vhp);

  attn_stats<<<dim3(SEQ / 128, BATCH), blk, 0, stream>>>(qhp, khp, mws, lws);
  attn_out<<<dim3(SEQ / 128, DM / 128, BATCH), blk, 0, stream>>>(qhp, khp, vhp, mws, lws, yp);

  gemm_xwt<_Float16, float><<<gg, blk, 0, stream>>>(yp, wp16, out);
}
